// ProductManifoldPhasorBlock_84825604096571
// MI455X (gfx1250) — matrix operand order repaired and checked by reading, not compiled, not run on hardware
//
#include <hip/hip_runtime.h>
#include <math.h>

// ---------------------------------------------------------------------------
// CDNA5 (gfx1250) WMMA types / helpers
// ---------------------------------------------------------------------------
typedef __attribute__((ext_vector_type(16))) __bf16 v16bf;
typedef __attribute__((ext_vector_type(8)))  __bf16 v8bf;
typedef __attribute__((ext_vector_type(8)))  float  v8f;

#define DIM    256
#define SEQL   2048
#define NBATCH 4
#define NTOK   (NBATCH*SEQL)   // 8192
#define PI_F   3.14159265358979f

__device__ __forceinline__ __bf16 f2bf(float f){
  unsigned u = __builtin_bit_cast(unsigned, f);
  u += 0x7fffu + ((u >> 16) & 1u);            // round to nearest even
  unsigned short s = (unsigned short)(u >> 16);
  return __builtin_bit_cast(__bf16, s);
}

__device__ __forceinline__ v8f wmma_bf(v16bf a, v16bf b, v8f c){
  // D(16x16,f32) = A(16x32,bf16) * B(32x16,bf16) + C
  return __builtin_amdgcn_wmma_f32_16x16x32_bf16(false, a, false, b, (short)0, c,
                                                 false, false);
}

__device__ __forceinline__ v16bf cat8(v8bf lo, v8bf hi){
  return __builtin_shufflevector(lo, hi, 0,1,2,3,4,5,6,7,8,9,10,11,12,13,14,15);
}

// A fragment: A is MxK row-major (bf16), lane holds row m=lane&15.
// Contiguous k order (k = 16*hi + e, as in the B loaders) = concat(row[k0+hi*16 .. +7], row[k0+hi*16+8 .. +15]) -> 2x16B loads.
__device__ __forceinline__ v16bf load_a(const __bf16* A, int ld, int k0, int lane){
  const int m = lane & 15, hi = (lane >> 4) & 1;
  const __bf16* row = A + m*ld + k0 + hi*16;
  v8bf lo = *(const v8bf*)(row);
  v8bf hh = *(const v8bf*)(row + 8);
  return cat8(lo, hh);
}

// B fragment from pre-packed weights: lane's 16 bf16 are contiguous (32B).
__device__ __forceinline__ v16bf load_b_packed(const __bf16* Wp, int ntiles,
                                               int kt, int nt, int lane){
  const __bf16* p = Wp + (((size_t)(kt*ntiles + nt))*32 + lane)*16;
  return *(const v16bf*)p;
}

// B fragment from transposed storage: B(k,n) = T[(n0+n)*ld + k0+k].
// Elements are contiguous -> single 32B vector load. ld must be mult of 16.
__device__ __forceinline__ v16bf load_bT(const __bf16* T, int ld, int k0, int n0, int lane){
  const int n = lane & 15, hi = (lane >> 4) & 1;
  const __bf16* p = T + (size_t)(n0 + n)*ld + k0 + hi*16;
  return *(const v16bf*)p;
}

// B fragment from the C/D-fragment-packed f32 state.
// Element e of the B fragment is feature k = kt*32 + hi*16 + e, column n.
// In packed-C order that is two contiguous v8f runs: [ft][tn][n] and [ft][tn][n+16].
__device__ __forceinline__ v16bf load_b_state(const v8f* Stp, int kt, int tn, int lane){
  const int n = lane & 15, hi = (lane >> 4) & 1;
  const int ft = 2*kt + hi;
  v8f r0 = Stp[(ft*4 + tn)*32 + n];
  v8f r1 = Stp[(ft*4 + tn)*32 + n + 16];
  v16bf b;
#pragma unroll
  for (int e = 0; e < 8; ++e){ b[e] = f2bf(r0[e]); b[e+8] = f2bf(r1[e]); }
  return b;
}

__device__ __forceinline__ v8f vzero8(){
  v8f z = {0.f,0.f,0.f,0.f,0.f,0.f,0.f,0.f};
  return z;
}

// ---------------------------------------------------------------------------
// Workspace layout (bytes). Weight regions hold fragment-packed bf16.
// ---------------------------------------------------------------------------
constexpr size_t OFF_XBF  = 0;                                  // 8192*256 bf16
constexpr size_t OFF_W1   = OFF_XBF  + (size_t)NTOK*DIM*2;      // 5 x 256x256 packed
constexpr size_t OFF_WKP2 = OFF_W1   + 5ull*256*256*2;          // 256x48 packed
constexpr size_t OFF_WQP2 = OFF_WKP2 + 256ull*48*2;
constexpr size_t OFF_WKH2 = OFF_WQP2 + 256ull*48*2;             // 256x32 packed
constexpr size_t OFF_WQH2 = OFF_WKH2 + 256ull*32*2;
constexpr size_t OFF_WOUT = OFF_WQH2 + 256ull*32*2;             // 256x256 packed
constexpr size_t OFF_Q2   = OFF_WOUT + 256ull*256*2;            // 8192x128 bf16
constexpr size_t OFF_K2   = OFF_Q2   + (size_t)NTOK*128*2;
constexpr size_t OFF_VBF  = OFF_K2   + (size_t)NTOK*128*2;      // 8192x256 bf16
constexpr size_t OFF_R    = OFF_VBF  + (size_t)NTOK*DIM*2;      // 8192x256 f32
constexpr size_t OFF_BC1  = OFF_R    + (size_t)NTOK*DIM*4;      // 1280 f32
constexpr size_t OFF_BC2  = OFF_BC1  + 1280ull*4;               // 160 f32

// ---------------------------------------------------------------------------
// fp32 -> bf16 conversion (x)
// ---------------------------------------------------------------------------
__global__ __launch_bounds__(256) void cvt_kernel(const float* __restrict__ src,
                                                  __bf16* __restrict__ dst, int n){
  const int i = blockIdx.x*256 + threadIdx.x;
  if (i < n) dst[i] = f2bf(src[i]);
}

// fp32 KxN row-major -> WMMA-B fragment-packed bf16.
__global__ __launch_bounds__(256) void pack_kernel(const float* __restrict__ W,
                                                   __bf16* __restrict__ dst,
                                                   int K, int N){
  const int i = blockIdx.x*256 + threadIdx.x;
  if (i >= K*N) return;
  const int e    = i & 15;
  const int lane = (i >> 4) & 31;
  const int tile = i >> 9;
  const int ntiles = N >> 4;
  const int nt = tile % ntiles;
  const int kt = tile / ntiles;
  const int k = kt*32 + ((lane >> 4) << 4) + e;
  const int n = nt*16 + (lane & 15);
  dst[i] = f2bf(W[k*N + n]);
}

// concatenated bias arrays (kills divergent pointer selects in stage1)
__global__ __launch_bounds__(256) void biascat1_kernel(
    const float* __restrict__ b0, const float* __restrict__ b1,
    const float* __restrict__ b2, const float* __restrict__ b3,
    const float* __restrict__ b4, float* __restrict__ dst){
  const int i = blockIdx.x*256 + threadIdx.x;
  if (i >= 1280) return;
  const int mat = i >> 8, c = i & 255;
  const float* s = (mat==0)?b0:(mat==1)?b1:(mat==2)?b2:(mat==3)?b3:b4;
  dst[i] = s[c];
}
__global__ __launch_bounds__(256) void biascat2_kernel(
    const float* __restrict__ kp, const float* __restrict__ qp,
    const float* __restrict__ kh, const float* __restrict__ qh,
    float* __restrict__ dst){
  const int i = threadIdx.x;
  if (i >= 160) return;
  dst[i] = (i < 48) ? kp[i] : (i < 96) ? qp[i-48] : (i < 128) ? kh[i-96] : qh[i-128];
}

// ---------------------------------------------------------------------------
// Stage 1: per-16-token tile, all MLPs + value proj + phasor math -> Q2/K2/V
// ---------------------------------------------------------------------------
__global__ __launch_bounds__(256) void stage1_kernel(
    const __bf16* __restrict__ Xbf, const __bf16* __restrict__ W1,
    const float* __restrict__ bias1,          // 1280 concat
    const __bf16* __restrict__ Wkp2, const __bf16* __restrict__ Wqp2,
    const __bf16* __restrict__ Wkh2, const __bf16* __restrict__ Wqh2,
    const float* __restrict__ bias2,          // 160 concat
    __bf16* __restrict__ Q2, __bf16* __restrict__ K2, __bf16* __restrict__ Vbf)
{
  __shared__ __bf16 Xs[16][256];   // token tile (bf16)
  __shared__ __bf16 Hs[16][1024];  // gelu(hidden) of kp1|qp1|kh1|qh1
  __shared__ __bf16 Vs16[16][256]; // V tile staged for coalesced store
  __shared__ float  Gs[16][160];   // kp2(48)|qp2(48)|kh2(32)|qh2(32)

  const int tid = threadIdx.x, lane = tid & 31, wave = tid >> 5;
  const int token0 = blockIdx.x * 16;

  for (int i = tid; i < 16*256; i += 256) Xs[i >> 8][i & 255] = Xbf[token0*256 + i];
  __syncthreads();

  // ---- first-layer GEMMs: X(16x256) x W(256x256), 5 matrices = 80 N-tiles
  for (int nt = wave; nt < 80; nt += 8){
    const int n0 = nt*16, mat = n0 >> 8, col0 = n0 & 255, ntl = col0 >> 4;
    const __bf16* W = W1 + (size_t)mat*256*256;
    __builtin_prefetch(W + ((size_t)ntl*32)*16, 0, 0);   // global_prefetch_b8
    v8f acc = vzero8();
#pragma unroll
    for (int kt = 0; kt < 8; ++kt){
      v16bf a = load_a(&Xs[0][0], 256, kt*32, lane);
      v16bf b = load_b_packed(W, 16, kt, ntl, lane);
      acc = wmma_bf(a, b, acc);
    }
    const int n = lane & 15, hi = lane >> 4;
    const float bv = bias1[n0 + n];
#pragma unroll
    for (int r = 0; r < 8; ++r){
      const int m = r + 8*hi;
      float v = acc[r] + bv;
      if (mat < 4){
        v = 0.5f * v * (1.0f + erff(v * 0.70710678118f));   // exact GELU
        Hs[m][mat*256 + col0 + n] = f2bf(v);
      } else {
        Vs16[m][col0 + n] = f2bf(v);
      }
    }
  }
  __syncthreads();

  // coalesced V writeback (8 contiguous bf16 = 16B per thread)
  for (int i = tid; i < 16*256/8; i += 256){
    *(v8bf*)(&Vbf[token0*256 + i*8]) = *(const v8bf*)(&Vs16[0][0] + i*8);
  }

  // ---- second-layer GEMMs: hidden(16x256) x W2 -> Gs (10 N-tiles)
  for (int nt = wave; nt < 10; nt += 8){
    const __bf16* W; int hb, col0, outc, ntiles;
    if (nt < 3)      { W = Wkp2; hb = 0; col0 = nt*16;     outc = col0;      ntiles = 3; }
    else if (nt < 6) { W = Wqp2; hb = 1; col0 = (nt-3)*16; outc = 48 + col0; ntiles = 3; }
    else if (nt < 8) { W = Wkh2; hb = 2; col0 = (nt-6)*16; outc = 96 + col0; ntiles = 2; }
    else             { W = Wqh2; hb = 3; col0 = (nt-8)*16; outc = 128+ col0; ntiles = 2; }
    const int ntl = col0 >> 4;
    v8f acc = vzero8();
#pragma unroll
    for (int kt = 0; kt < 8; ++kt){
      v16bf a = load_a(&Hs[0][hb*256], 1024, kt*32, lane);
      v16bf b = load_b_packed(W, ntiles, kt, ntl, lane);
      acc = wmma_bf(a, b, acc);
    }
    const int n = lane & 15, hi = lane >> 4;
    const float bv = bias2[outc + n];
#pragma unroll
    for (int r = 0; r < 8; ++r) Gs[r + 8*hi][outc + n] = acc[r] + bv;
  }
  __syncthreads();

  // ---- phasor math -> K2/Q2 features [re(64) | im(64)]
  const int tok = tid >> 4;
  for (int o = tid & 15; o < 64; o += 16){
    float kr, ki, qr, qi;
    if (o < 48){
      const float kp = tanhf(Gs[tok][o])      * PI_F;
      const float qp = tanhf(Gs[tok][48 + o]) * PI_F;
      sincosf(kp, &ki, &kr);
      sincosf(qp, &qi, &qr);
    } else {
      const int j = o - 48;
      float a0 = tanhf(Gs[tok][96 + 2*j])   * 0.9f;
      float a1 = tanhf(Gs[tok][96 + 2*j+1]) * 0.9f;
      float ang = atan2f(a1, a0);
      float amp = 1.0f - sqrtf(a0*a0 + a1*a1);
      float s, c; sincosf(ang, &s, &c);
      kr = amp * c; ki = amp * s;
      a0 = tanhf(Gs[tok][128 + 2*j])   * 0.9f;
      a1 = tanhf(Gs[tok][128 + 2*j+1]) * 0.9f;
      ang = atan2f(a1, a0);
      amp = 1.0f - sqrtf(a0*a0 + a1*a1);
      sincosf(ang, &s, &c);
      qr = amp * c; qi = amp * s;
    }
    const int t = token0 + tok;
    K2[t*128 + o]      = f2bf(kr);  K2[t*128 + 64 + o] = f2bf(ki);
    Q2[t*128 + o]      = f2bf(qr);  Q2[t*128 + 64 + o] = f2bf(qi);
  }
}

// ---------------------------------------------------------------------------
// Causal linear attention: chunked scan, chunk=32, f32 KV state in LDS
// (state kept in C/D-fragment-packed order -> vector LDS accesses).
// Grid: 4 batches x 4 value-dim slices of 64.
// ---------------------------------------------------------------------------
__global__ __launch_bounds__(256) void attn_kernel(
    const __bf16* __restrict__ Q2, const __bf16* __restrict__ K2,
    const __bf16* __restrict__ Vbf, float* __restrict__ R)
{
  __shared__ __bf16 Qs[32][128];
  __shared__ __bf16 Ks[32][128];    // token-major (scores B operand)
  __shared__ __bf16 KsT[128][32];   // feat-major  (state-update A operand)
  __shared__ __bf16 VsT[64][32];    // V transposed: contiguous B fragments
  __shared__ __bf16 Sc[32][32];     // masked scores (bf16)
  __shared__ v8f    Stp[8*4*32];    // KV state f32, fragment-packed [ft][dt][lane]

  const int tid = threadIdx.x, lane = tid & 31, wave = tid >> 5;
  const int b = blockIdx.x >> 2, d0 = (blockIdx.x & 3) * 64;

  for (int i = tid; i < 8*4*32; i += 256) Stp[i] = vzero8();
  __syncthreads();

  for (int c = 0; c < 64; ++c){
    const int tb = c*32;
    const int gt0 = b*SEQL + tb;
    for (int i = tid; i < 32*128; i += 256){
      const int t = i >> 7, f = i & 127;       // f fastest: coalesced global
      Qs[t][f] = Q2[(gt0 + t)*128 + f];
      const __bf16 kv = K2[(gt0 + t)*128 + f];
      Ks[t][f]  = kv;
      KsT[f][t] = kv;
    }
    for (int i = tid; i < 32*64; i += 256){
      const int t = i >> 6, d = i & 63;        // d fastest: coalesced global
      VsT[d][t] = Vbf[(gt0 + t)*256 + d0 + d];
    }
    __syncthreads();

    // ---- scores = tril(Q Kt), 2x2 tiles (waves 0..3)
    if (wave < 4){
      const int tm = wave >> 1, tn = wave & 1;
      const int n = lane & 15, hi = lane >> 4;
      if (tn <= tm){
        v8f acc = vzero8();
#pragma unroll
        for (int kt = 0; kt < 4; ++kt){
          v16bf a  = load_a (&Qs[tm*16][0], 128, kt*32, lane);
          v16bf bt = load_bT(&Ks[0][0],     128, kt*32, tn*16, lane);
          acc = wmma_bf(a, bt, acc);
        }
#pragma unroll
        for (int r = 0; r < 8; ++r){
          const int m = tm*16 + r + 8*hi, j = tn*16 + n;
          Sc[m][j] = f2bf((j <= m) ? acc[r] : 0.f);
        }
      } else {
#pragma unroll
        for (int r = 0; r < 8; ++r) Sc[tm*16 + r + 8*hi][tn*16 + n] = f2bf(0.f);
      }
    }
    __syncthreads();

    // ---- out = Q*S_prev + Sc*V   (2x4 tiles, one per wave)
    {
      const int tm = wave >> 2, tn = wave & 3;
      v8f acc = vzero8();
#pragma unroll
      for (int kt = 0; kt < 4; ++kt){
        v16bf a  = load_a(&Qs[tm*16][0], 128, kt*32, lane);
        v16bf bs = load_b_state(Stp, kt, tn, lane);
        acc = wmma_bf(a, bs, acc);
      }
      {
        v16bf a  = load_a (&Sc[tm*16][0], 32, 0, lane);
        v16bf bv = load_bT(&VsT[0][0],    32, 0, tn*16, lane);
        acc = wmma_bf(a, bv, acc);
      }
      const int n = lane & 15, hi = lane >> 4;
#pragma unroll
      for (int r = 0; r < 8; ++r){
        const int m = tm*16 + r + 8*hi;
        const int t = tb + m;
        const float sc = rsqrtf((float)(t + 1) * 64.0f);
        R[(b*SEQL + t)*256 + d0 + tn*16 + n] = acc[r] * sc;
      }
    }
    __syncthreads();   // all reads of Stp done before update

    // ---- S_state += Kt * V  (8x4 tiles, 4 per wave), f32 accumulate via C
    for (int t2 = wave; t2 < 32; t2 += 8){
      const int tm = t2 >> 2, tn = t2 & 3;
      v16bf a  = load_a (&KsT[tm*16][0], 32, 0, lane);   // A = K^T (feat x token)
      v16bf bv = load_bT(&VsT[0][0],     32, 0, tn*16, lane);
      v8f   cc = Stp[(tm*4 + tn)*32 + lane];
      cc = wmma_bf(a, bv, cc);
      Stp[(tm*4 + tn)*32 + lane] = cc;
    }
    __syncthreads();
  }
}

// ---------------------------------------------------------------------------
// LayerNorm + out projection + residual
// ---------------------------------------------------------------------------
__global__ __launch_bounds__(256) void final_kernel(
    const float* __restrict__ R, const float* __restrict__ x,
    const float* __restrict__ ln_g, const float* __restrict__ ln_b,
    const __bf16* __restrict__ Wout, const float* __restrict__ b_out,
    float* __restrict__ out)
{
  __shared__ float  Rs[16][256];
  __shared__ __bf16 Ls[16][256];
  __shared__ float  red1[16][16], red2[16][16];
  __shared__ float  muA[16], rsA[16];

  const int tid = threadIdx.x, lane = tid & 31, wave = tid >> 5;
  const int token0 = blockIdx.x * 16;

  for (int i = tid; i < 16*256; i += 256) Rs[i >> 8][i & 255] = R[token0*256 + i];
  __syncthreads();

  const int tok = tid >> 4, part = tid & 15;
  {
    float s = 0.f, s2 = 0.f;
#pragma unroll
    for (int j = 0; j < 16; ++j){
      const float v = Rs[tok][part*16 + j];
      s += v; s2 += v*v;
    }
    red1[tok][part] = s; red2[tok][part] = s2;
  }
  __syncthreads();
  if (part == 0){
    float S = 0.f, S2 = 0.f;
#pragma unroll
    for (int j = 0; j < 16; ++j){ S += red1[tok][j]; S2 += red2[tok][j]; }
    const float mu = S * (1.0f/256.0f);
    muA[tok] = mu;
    rsA[tok] = rsqrtf(S2 * (1.0f/256.0f) - mu*mu + 1e-5f);
  }
  __syncthreads();
  {
    const float mu = muA[tok], rs = rsA[tok];
#pragma unroll
    for (int j = 0; j < 16; ++j){
      const int col = part*16 + j;
      Ls[tok][col] = f2bf((Rs[tok][col] - mu) * rs * ln_g[col] + ln_b[col]);
    }
  }
  __syncthreads();

  for (int nt = wave; nt < 16; nt += 8){
    const int n0 = nt*16;
    v8f acc = vzero8();
#pragma unroll
    for (int kt = 0; kt < 8; ++kt){
      v16bf a = load_a(&Ls[0][0], 256, kt*32, lane);
      v16bf b = load_b_packed(Wout, 16, kt, n0 >> 4, lane);
      acc = wmma_bf(a, b, acc);
    }
    const int n = lane & 15, hi = lane >> 4;
#pragma unroll
    for (int r = 0; r < 8; ++r){
      const int m = r + 8*hi, col = n0 + n, t = token0 + m;
      out[t*256 + col] = x[t*256 + col] + acc[r] + b_out[col];
    }
  }
}

// ---------------------------------------------------------------------------
// launch
// ---------------------------------------------------------------------------
extern "C" void kernel_launch(void* const* d_in, const int* in_sizes, int n_in,
                              void* d_out, int out_size, void* d_ws, size_t ws_size,
                              hipStream_t stream) {
  (void)in_sizes; (void)n_in; (void)out_size; (void)ws_size;
  char* ws = (char*)d_ws;
  const float* x = (const float*)d_in[0];
  // d_in order: x, kp1(W,b), kp2, qp1, qp2, kh1, kh2, qh1, qh2, val, out, ln_g, ln_b
  cvt_kernel<<<(NTOK*DIM + 255)/256, 256, 0, stream>>>(
      (const float*)d_in[0], (__bf16*)(ws + OFF_XBF), NTOK*DIM);

  auto pack = [&](int idx, size_t off, int K, int N){
    pack_kernel<<<(K*N + 255)/256, 256, 0, stream>>>((const float*)d_in[idx],
                                                     (__bf16*)(ws + off), K, N);
  };
  pack(1,  OFF_W1 + 0ull*256*256*2, 256, 256);   // kp1
  pack(5,  OFF_W1 + 1ull*256*256*2, 256, 256);   // qp1
  pack(9,  OFF_W1 + 2ull*256*256*2, 256, 256);   // kh1
  pack(13, OFF_W1 + 3ull*256*256*2, 256, 256);   // qh1
  pack(17, OFF_W1 + 4ull*256*256*2, 256, 256);   // val
  pack(3,  OFF_WKP2, 256, 48);
  pack(7,  OFF_WQP2, 256, 48);
  pack(11, OFF_WKH2, 256, 32);
  pack(15, OFF_WQH2, 256, 32);
  pack(19, OFF_WOUT, 256, 256);

  biascat1_kernel<<<5, 256, 0, stream>>>(
      (const float*)d_in[2],  (const float*)d_in[6],  (const float*)d_in[10],
      (const float*)d_in[14], (const float*)d_in[18], (float*)(ws + OFF_BC1));
  biascat2_kernel<<<1, 256, 0, stream>>>(
      (const float*)d_in[4],  (const float*)d_in[8],  (const float*)d_in[12],
      (const float*)d_in[16], (float*)(ws + OFF_BC2));

  stage1_kernel<<<NTOK/16, 256, 0, stream>>>(
      (const __bf16*)(ws + OFF_XBF), (const __bf16*)(ws + OFF_W1),
      (const float*)(ws + OFF_BC1),
      (const __bf16*)(ws + OFF_WKP2), (const __bf16*)(ws + OFF_WQP2),
      (const __bf16*)(ws + OFF_WKH2), (const __bf16*)(ws + OFF_WQH2),
      (const float*)(ws + OFF_BC2),
      (__bf16*)(ws + OFF_Q2), (__bf16*)(ws + OFF_K2), (__bf16*)(ws + OFF_VBF));

  attn_kernel<<<NBATCH*4, 256, 0, stream>>>(
      (const __bf16*)(ws + OFF_Q2), (const __bf16*)(ws + OFF_K2),
      (const __bf16*)(ws + OFF_VBF), (float*)(ws + OFF_R));

  final_kernel<<<NTOK/16, 256, 0, stream>>>(
      (const float*)(ws + OFF_R), x,
      (const float*)d_in[21], (const float*)d_in[22],
      (const __bf16*)(ws + OFF_WOUT), (const float*)d_in[20],
      (float*)d_out);
}
